// VoiceSynthesizerModel_11132555231573
// MI455X (gfx1250) — matrix operand order repaired and checked by reading, not compiled, not run on hardware
//
#include <hip/hip_runtime.h>
#include <hip/hip_bf16.h>
#include <math.h>

typedef __bf16 bf16;
typedef __attribute__((ext_vector_type(16))) __bf16 v16bf;
typedef __attribute__((ext_vector_type(8)))  __bf16 v8bf;
typedef __attribute__((ext_vector_type(8)))  float  v8f;

#define DEV static __device__ __forceinline__
#define CDIV(a,b) (((a)+(b)-1)/(b))

// ---- model dims ----
#define BATCH 16
#define TTEXT 200
#define LMEL  300
#define EMB   256
#define ENCH  512
#define DECH  512
#define NMELS 80
#define ATTND 128
#define ROWS_ENC 216   // padded time rows per batch, encoder convs
#define ROWS_PN  312   // padded time rows per batch, postnet convs

DEV bf16 f2bf(float f){ return (bf16)f; }
DEV float bf2f(bf16 h){ return (float)h; }
DEV float sigmf(float x){ return 1.0f/(1.0f+__expf(-x)); }

// ---------------- WMMA fragment helpers ----------------
// A: row-major bf16 (lda halves/row). ISA 16-bit A 16x32 layout:
// lane 0-15: m=lane, halves -> K {kb+0..7, kb+16..23}; lane 16-31: K {kb+8..15, kb+24..31}
DEV v16bf load_a_frag(const bf16* A, int lda, int mbase, int kbase){
  int lane = threadIdx.x & 31;
  int row  = mbase + (lane & 15);
  int k0   = kbase + ((lane >> 4) << 3);
  const bf16* p = A + (size_t)row * lda + k0;
  v8bf lo = *(const v8bf*)p;
  v8bf hi = *(const v8bf*)(p + 16);
  v16bf r;
#pragma unroll
  for (int i = 0; i < 8; i++){ r[i] = lo[i]; r[8+i] = hi[i]; }
  return r;
}
// B: pre-swizzled tiles, tile id = kt*ntiles+nt, 512 halves/tile, lane-contiguous.
DEV v16bf load_b_frag(const bf16* Bsw, int ntiles, int kt, int nt){
  int lane = threadIdx.x & 31;
  return *(const v16bf*)(Bsw + (((size_t)kt * ntiles + nt) << 9) + lane * 16);
}
DEV v8f wmma_bf16(v16bf a, v16bf b, v8f c){
  return __builtin_amdgcn_wmma_f32_16x16x32_bf16(false, a, false, b, (short)0, c, false, false);
}

// ---------------- weight conversion: f32 strided -> bf16 swizzled B tiles ----------------
__global__ void k_convert_b(const float* __restrict__ src, bf16* __restrict__ dst,
                            int N, int K, int sN, int sK, int off, int Kp){
  int ntiles = N >> 4;
  size_t total = (size_t)(Kp >> 5) * ntiles * 512;
  for (size_t idx = (size_t)blockIdx.x * blockDim.x + threadIdx.x; idx < total;
       idx += (size_t)gridDim.x * blockDim.x){
    int t = (int)(idx >> 9), w = (int)(idx & 511), l = w >> 4, h = w & 15;
    int nt = t % ntiles, kt = t / ntiles;
    int k = kt * 32 + ((l >> 4) << 3) + (h & 7) + ((h >> 3) << 4);
    int n = nt * 16 + (l & 15);
    float v = (k < K && n < N) ? src[(size_t)off + (size_t)n * sN + (size_t)k * sK] : 0.0f;
    dst[idx] = f2bf(v);
  }
}

// ---------------- generic GEMM with NB-way N-tile blocking (A reuse) ----------------
template<int NB>
__global__ void k_gemm(const bf16* __restrict__ A, int lda, const bf16* __restrict__ Bsw,
                       const float* __restrict__ bias,
                       float* __restrict__ Cf, bf16* __restrict__ Cb,
                       int M, int N, int K, int relu){
  int Mt = M >> 4, Nt = N >> 4, Kt = K >> 5;
  int ng = Nt / NB;
  int wave = (blockIdx.x * blockDim.x + threadIdx.x) >> 5;
  int nw = (gridDim.x * blockDim.x) >> 5;
  int lane = threadIdx.x & 31;
  int total = Mt * ng;
  for (int tid = wave; tid < total; tid += nw){
    int mt = tid / ng, nt0 = (tid % ng) * NB;
    v8f acc[NB];
#pragma unroll
    for (int u = 0; u < NB; u++)
#pragma unroll
      for (int r = 0; r < 8; r++) acc[u][r] = 0.f;
    for (int kt = 0; kt < Kt; kt++){
      v16bf a = load_a_frag(A, lda, mt * 16, kt * 32);
      if (kt + 1 < Kt)
        __builtin_prefetch(Bsw + (((size_t)(kt + 1) * Nt + nt0) << 9), 0, 0);
#pragma unroll
      for (int u = 0; u < NB; u++)
        acc[u] = wmma_bf16(a, load_b_frag(Bsw, Nt, kt, nt0 + u), acc[u]);
    }
    int rbase = mt * 16 + ((lane >> 4) << 3);
#pragma unroll
    for (int u = 0; u < NB; u++){
      int col = (nt0 + u) * 16 + (lane & 15);
      float bv = bias ? bias[col] : 0.f;
#pragma unroll
      for (int r = 0; r < 8; r++){
        float v = acc[u][r] + bv;
        if (relu) v = fmaxf(v, 0.f);
        size_t o = (size_t)(rbase + r) * N + col;
        if (Cf) Cf[o] = v;
        if (Cb) Cb[o] = f2bf(v);
      }
    }
  }
}

// ---------------- K=5 conv as 5 tap-GEMMs + (optional) BN + ReLU ----------------
template<int NB>
__global__ void k_conv5(const bf16* __restrict__ Xpad, int rowsPerB, int ldx,
                        const bf16* __restrict__ Wsw, int Kp, int Coutp, int Cout,
                        const float* __restrict__ cb, const float* __restrict__ g,
                        const float* __restrict__ be,
                        bf16* __restrict__ Ypad, int ldy, float* __restrict__ Yres,
                        int Tvalid, int Ttiles, int relu){
  int Kt = Kp >> 5, Nt = Coutp >> 4;
  int ng = Nt / NB;
  int wave = (blockIdx.x * blockDim.x + threadIdx.x) >> 5;
  int nw = (gridDim.x * blockDim.x) >> 5;
  int lane = threadIdx.x & 31;
  int total = BATCH * Ttiles * ng;
  for (int tid = wave; tid < total; tid += nw){
    int nt0 = (tid % ng) * NB;
    int tt = (tid / ng) % Ttiles;
    int b  = tid / (ng * Ttiles);
    v8f acc[NB];
#pragma unroll
    for (int u = 0; u < NB; u++)
#pragma unroll
      for (int r = 0; r < 8; r++) acc[u][r] = 0.f;
    for (int tap = 0; tap < 5; tap++){
      int mbase = b * rowsPerB + tt * 16 + tap;
      const bf16* Wt = Wsw + (((size_t)tap * Kt * Nt) << 9);
      for (int kt = 0; kt < Kt; kt++){
        v16bf a = load_a_frag(Xpad, ldx, mbase, kt * 32);
        if (kt + 1 < Kt)
          __builtin_prefetch(Wt + (((size_t)(kt + 1) * Nt + nt0) << 9), 0, 0);
#pragma unroll
        for (int u = 0; u < NB; u++)
          acc[u] = wmma_bf16(a, load_b_frag(Wt, Nt, kt, nt0 + u), acc[u]);
      }
    }
#pragma unroll
    for (int u = 0; u < NB; u++){
      int c = (nt0 + u) * 16 + (lane & 15);
      if (c >= Cout) continue;
      float s, b2;
      if (g){ s = g[c] * rsqrtf(1.0f + 1e-5f); b2 = cb[c] * s + be[c]; }
      else  { s = 1.0f; b2 = cb[c]; }
#pragma unroll
      for (int r = 0; r < 8; r++){
        int t = tt * 16 + ((lane >> 4) << 3) + r;
        if (t >= Tvalid) continue;
        float v = acc[u][r] * s + b2;
        if (relu) v = fmaxf(v, 0.f);
        if (Ypad) Ypad[(size_t)(b * rowsPerB + t + 2) * ldy + c] = f2bf(v);
        if (Yres){
          size_t o = (size_t)b * Cout * Tvalid + (size_t)c * Tvalid + t;
          Yres[o] = Yres[o] + v;
        }
      }
    }
  }
}

// ---------------- small utility kernels ----------------
__global__ void k_embed(const int* __restrict__ text, const float* __restrict__ emb,
                        bf16* __restrict__ xpad){
  int idx = blockIdx.x * blockDim.x + threadIdx.x;
  if (idx >= BATCH * TTEXT * EMB) return;
  int c = idx % EMB, t = (idx / EMB) % TTEXT, b = idx / (EMB * TTEXT);
  int tok = text[b * TTEXT + t];
  xpad[(size_t)(b * ROWS_ENC + t + 2) * EMB + c] = f2bf(emb[(size_t)tok * EMB + c]);
}

// repack conv layout -> LSTM row order, staged through LDS with CDNA5 async copy.
// Each thread owns a private 16B LDS slot: per-lane async gather, wait, b128 store out.
__global__ void k_repack(const bf16* __restrict__ xpad, bf16* __restrict__ xsA){
  __shared__ char stage[256 * 16];
  int tid = threadIdx.x;
  unsigned lds_off = (unsigned)(size_t)(&stage[tid * 16]);
  for (int gch = blockIdx.x * blockDim.x + tid; gch < TTEXT * BATCH * 32;
       gch += gridDim.x * blockDim.x){
    int c = gch & 31, row = gch >> 5;          // 32 x 16B chunks per 256-col row
    int t = row / BATCH, b = row & (BATCH - 1);
    const bf16* src = xpad + (size_t)(b * ROWS_ENC + t + 2) * EMB + c * 8;
    unsigned long long ga = (unsigned long long)(size_t)src;
    asm volatile("global_load_async_to_lds_b128 %0, %1, off"
                 :: "v"(lds_off), "v"(ga) : "memory");
    asm volatile("s_wait_asynccnt 0x0" ::: "memory");
    uint4 v = *(const uint4*)&stage[tid * 16];
    *(uint4*)(xsA + (size_t)gch * 8) = v;
  }
}

__global__ void k_vecadd(const float* a, const float* b, float* o, int n){
  int i = blockIdx.x * blockDim.x + threadIdx.x;
  if (i < n) o[i] = a[i] + b[i];
}

__global__ void k_prev(const float* __restrict__ mel, bf16* __restrict__ prevA){
  int idx = blockIdx.x * blockDim.x + threadIdx.x;
  if (idx >= LMEL * BATCH * 96) return;
  int c = idx % 96, row = idx / 96;
  int step = row / BATCH, b = row % BATCH;
  float v = 0.f;
  if (c < NMELS && step > 0) v = mel[(size_t)b * NMELS * LMEL + (size_t)c * LMEL + (step - 1)];
  prevA[idx] = f2bf(v);
}

__global__ void k_m2(const float* __restrict__ Wl, const float* __restrict__ locw,
                     const float* __restrict__ locb, float* __restrict__ M2f,
                     float* __restrict__ cbias){
  int idx = blockIdx.x * blockDim.x + threadIdx.x;
  if (idx < 128 * 31){
    int a = idx / 31, k = idx % 31;
    float s = 0.f;
    for (int f = 0; f < 32; f++) s += Wl[a * 32 + f] * locw[f * 31 + k];
    M2f[a * 31 + k] = s;
  }
  if (idx < 128){
    float s = 0.f;
    for (int f = 0; f < 32; f++) s += Wl[idx * 32 + f] * locb[f];
    cbias[idx] = s;
  }
}

// ---------------- persistent BiLSTM encoder (block 0 = fwd, block 1 = bwd) ----------------
__global__ __launch_bounds__(1024) void k_lstm_enc(const float* __restrict__ xprojF,
                                                   const float* __restrict__ xprojB,
                                                   const bf16* __restrict__ BhhF,
                                                   const bf16* __restrict__ BhhB,
                                                   bf16* __restrict__ encA){
  __shared__ bf16 hs[16][512];
  const int dir = blockIdx.x;
  const float* xproj = dir ? xprojB : xprojF;
  const bf16* Bhh = dir ? BhhB : BhhF;
  int tid = threadIdx.x, lane = tid & 31, W = tid >> 5;
  int jl = lane & 15, j = W * 16 + jl, bb = ((lane >> 4) << 3);
  for (int i = tid; i < 16 * 512; i += 1024) ((bf16*)hs)[i] = f2bf(0.f);
  v8f c;
#pragma unroll
  for (int r = 0; r < 8; r++) c[r] = 0.f;
  __syncthreads();
  for (int s = 0; s < TTEXT; s++){
    int t = dir ? (TTEXT - 1 - s) : s;
    v8f acc[4];
#pragma unroll
    for (int g = 0; g < 4; g++)
#pragma unroll
      for (int r = 0; r < 8; r++)
        acc[g][r] = xproj[(size_t)(t * 16 + bb + r) * 2048 + g * 512 + j];
    for (int kt = 0; kt < 16; kt++){
      v16bf a = load_a_frag(&hs[0][0], 512, 0, kt * 32);
#pragma unroll
      for (int g = 0; g < 4; g++)
        acc[g] = wmma_bf16(a, load_b_frag(Bhh, 128, kt, g * 32 + W), acc[g]);
    }
    float hh[8];
#pragma unroll
    for (int r = 0; r < 8; r++){
      float gi = sigmf(acc[0][r]);
      float gf = sigmf(acc[1][r]);
      float gg = tanhf(acc[2][r]);
      float go = sigmf(acc[3][r]);
      c[r] = gf * c[r] + gi * gg;
      hh[r] = go * tanhf(c[r]);
    }
    __syncthreads();
#pragma unroll
    for (int r = 0; r < 8; r++){
      int b = bb + r;
      bf16 hv = f2bf(hh[r]);
      hs[b][j] = hv;
      encA[(size_t)(b * TTEXT + t) * 1024 + dir * 512 + j] = hv;
    }
    __syncthreads();
  }
}

// ---------------- persistent attention decoder (single workgroup, 300 steps) ----------------
__global__ __launch_bounds__(1024) void k_decoder(
    const float* __restrict__ preproj,   // (L*16, 2048) = pre@WihPre + bih + bhh
    const bf16* __restrict__ BattCtx,    // K512 N2048
    const bf16* __restrict__ BattHh,     // K512 N2048
    const bf16* __restrict__ BdecIh,     // K1024 N2048
    const bf16* __restrict__ BdecHh,     // K512 N2048
    const float* __restrict__ biasDec,   // 2048 (bih+bhh)
    const bf16* __restrict__ BWq,        // K512 N128
    const bf16* __restrict__ BM2,        // K32 N128 (Wl*loc_w folded)
    const float* __restrict__ cbias,     // 128 (Wl*loc_b)
    const float* __restrict__ keys,      // (3200,128)
    const float* __restrict__ wscore,    // 128
    const float* __restrict__ encV,      // (3200,512) enc@Wv
    const bf16* __restrict__ BWmel,      // K512 N80
    const float* __restrict__ bmel,      // 80
    const float* __restrict__ Wstop, const float* __restrict__ bstop,
    bf16* __restrict__ ctxG,             // (16,512) bf16 scratch (pre-zeroed)
    bf16* __restrict__ pmel,             // postnet padded input (16*312,96)
    float* __restrict__ outMel, float* __restrict__ outStop, float* __restrict__ outAttn){
  __shared__ bf16 h1s[16][512];
  __shared__ bf16 h2s[16][512];
  __shared__ bf16 awp[16][240];          // aw window: index i holds aw[i-15], pads zero
  __shared__ float eS[16][200];          // scores -> softmax in place
  __shared__ bf16 qS[16][128];
  int tid = threadIdx.x, lane = tid & 31, W = tid >> 5;
  int jl = lane & 15, j = W * 16 + jl, bb = ((lane >> 4) << 3);
  for (int i = tid; i < 16 * 512; i += 1024){ ((bf16*)h1s)[i] = f2bf(0.f); ((bf16*)h2s)[i] = f2bf(0.f); }
  for (int i = tid; i < 16 * 240; i += 1024) ((bf16*)awp)[i] = f2bf(0.f);
  v8f c1, c2;
#pragma unroll
  for (int r = 0; r < 8; r++){ c1[r] = 0.f; c2[r] = 0.f; }
  __syncthreads();

  for (int step = 0; step < LMEL; step++){
    // ---- Phase A: att_lstm: z = preproj[step] + ctx@WihCtx + h1@Whh ----
    v8f acc[4];
#pragma unroll
    for (int g = 0; g < 4; g++)
#pragma unroll
      for (int r = 0; r < 8; r++)
        acc[g][r] = preproj[(size_t)(step * 16 + bb + r) * 2048 + g * 512 + j];
    for (int kt = 0; kt < 16; kt++){
      v16bf a = load_a_frag(&h1s[0][0], 512, 0, kt * 32);
#pragma unroll
      for (int g = 0; g < 4; g++)
        acc[g] = wmma_bf16(a, load_b_frag(BattHh, 128, kt, g * 32 + W), acc[g]);
    }
    for (int kt = 0; kt < 16; kt++){
      v16bf a = load_a_frag(ctxG, 512, 0, kt * 32);
#pragma unroll
      for (int g = 0; g < 4; g++)
        acc[g] = wmma_bf16(a, load_b_frag(BattCtx, 128, kt, g * 32 + W), acc[g]);
    }
    float hh[8];
#pragma unroll
    for (int r = 0; r < 8; r++){
      float gi = sigmf(acc[0][r]), gf = sigmf(acc[1][r]);
      float gg = tanhf(acc[2][r]), go = sigmf(acc[3][r]);
      c1[r] = gf * c1[r] + gi * gg;
      hh[r] = go * tanhf(c1[r]);
    }
    __syncthreads();
#pragma unroll
    for (int r = 0; r < 8; r++) h1s[bb + r][j] = f2bf(hh[r]);
    __syncthreads();

    // ---- Phase B: waves 0-7: q = h1@Wq ; waves 8-31: zero score buffer ----
    if (W < 8){
      v8f q;
#pragma unroll
      for (int r = 0; r < 8; r++) q[r] = 0.f;
      for (int kt = 0; kt < 16; kt++)
        q = wmma_bf16(load_a_frag(&h1s[0][0], 512, 0, kt * 32), load_b_frag(BWq, 8, kt, W), q);
#pragma unroll
      for (int r = 0; r < 8; r++) qS[bb + r][W * 16 + jl] = f2bf(q[r]);
    } else {
      for (int i = tid - 256; i < 16 * 200; i += 768) ((float*)eS)[i] = 0.f;
    }
    __syncthreads();

    // ---- Phase C: location features (31-tap conv folded into K=32 GEMM) + fused tanh/score ----
    for (int job = W; job < 208 * 8; job += 32){
      int mt = job >> 3, nt = job & 7;
      int b = mt / 13, tt = mt % 13;
      int a_col = nt * 16 + jl;
      int basei = tt * 16 + jl + ((lane >> 4) << 3);
      v16bf a;
#pragma unroll
      for (int h = 0; h < 8; h++){ a[h] = awp[b][basei + h]; a[8 + h] = awp[b][basei + 16 + h]; }
      v8f ac;
      float cb0 = cbias[a_col];
#pragma unroll
      for (int r = 0; r < 8; r++) ac[r] = cb0;
      ac = wmma_bf16(a, load_b_frag(BM2, 8, 0, nt), ac);
      float qv = bf2f(qS[b][a_col]);
      float ws = wscore[a_col];
#pragma unroll
      for (int r = 0; r < 8; r++){
        int t = tt * 16 + ((lane >> 4) << 3) + r;
        float v = 0.f;
        if (t < 200)
          v = tanhf(ac[r] + qv + keys[(size_t)(b * 200 + t) * 128 + a_col]) * ws;
#pragma unroll
        for (int m = 1; m < 16; m <<= 1) v += __shfl_xor(v, m, 32);
        if (jl == 0 && t < 200) atomicAdd(&eS[b][t], v);
      }
    }
    __syncthreads();

    // ---- Phase D: softmax over Tt per batch (waves 0-15) ----
    if (W < 16){
      int b = W;
      float mx = -1e30f;
      for (int t = lane; t < 200; t += 32) mx = fmaxf(mx, eS[b][t]);
#pragma unroll
      for (int m = 16; m >= 1; m >>= 1) mx = fmaxf(mx, __shfl_xor(mx, m, 32));
      float sum = 0.f;
      for (int t = lane; t < 200; t += 32){ float ev = __expf(eS[b][t] - mx); eS[b][t] = ev; sum += ev; }
#pragma unroll
      for (int m = 16; m >= 1; m >>= 1) sum += __shfl_xor(sum, m, 32);
      float inv = 1.f / sum;
      for (int t = lane; t < 200; t += 32){
        float av = eS[b][t] * inv;
        eS[b][t] = av;
        awp[b][t + 15] = f2bf(av);
        outAttn[(size_t)b * LMEL * TTEXT + (size_t)step * TTEXT + t] = av;
      }
    }
    __syncthreads();

    // ---- Phase E: ctx = (aw @ encV)  (encV = enc@Wv precomputed) ----
    for (int idx = tid; idx < 16 * 512; idx += 1024){
      int b = idx >> 9, d = idx & 511;
      float s = 0.f;
      const float* ev = encV + (size_t)b * 200 * 512 + d;
#pragma unroll 4
      for (int t = 0; t < 200; t++) s += eS[b][t] * ev[(size_t)t * 512];
      ctxG[b * 512 + d] = f2bf(s);
    }
    __threadfence_block();
    __syncthreads();

    // ---- Phase F: dec_lstm: z = bias + [h1,ctx]@Wih + h2@Whh ----
    v8f ad[4];
#pragma unroll
    for (int g = 0; g < 4; g++){
      float bv = biasDec[g * 512 + j];
#pragma unroll
      for (int r = 0; r < 8; r++) ad[g][r] = bv;
    }
    for (int kt = 0; kt < 16; kt++){
      v16bf a = load_a_frag(&h1s[0][0], 512, 0, kt * 32);
#pragma unroll
      for (int g = 0; g < 4; g++)
        ad[g] = wmma_bf16(a, load_b_frag(BdecIh, 128, kt, g * 32 + W), ad[g]);
    }
    for (int kt = 0; kt < 16; kt++){
      v16bf a = load_a_frag(ctxG, 512, 0, kt * 32);
#pragma unroll
      for (int g = 0; g < 4; g++)
        ad[g] = wmma_bf16(a, load_b_frag(BdecIh, 128, 16 + kt, g * 32 + W), ad[g]);
    }
    for (int kt = 0; kt < 16; kt++){
      v16bf a = load_a_frag(&h2s[0][0], 512, 0, kt * 32);
#pragma unroll
      for (int g = 0; g < 4; g++)
        ad[g] = wmma_bf16(a, load_b_frag(BdecHh, 128, kt, g * 32 + W), ad[g]);
    }
#pragma unroll
    for (int r = 0; r < 8; r++){
      float gi = sigmf(ad[0][r]), gf = sigmf(ad[1][r]);
      float gg = tanhf(ad[2][r]), go = sigmf(ad[3][r]);
      c2[r] = gf * c2[r] + gi * gg;
      hh[r] = go * tanhf(c2[r]);
    }
    __syncthreads();
#pragma unroll
    for (int r = 0; r < 8; r++) h2s[bb + r][j] = f2bf(hh[r]);
    __syncthreads();

    // ---- Phase G: mel = h2@Wmel + bmel ; stop = sigmoid(h2.Wstop+b) ----
    if (W < 5){
      v8f am;
      int col = W * 16 + jl;
      float bm = (col < NMELS) ? bmel[col] : 0.f;
#pragma unroll
      for (int r = 0; r < 8; r++) am[r] = bm;
      for (int kt = 0; kt < 16; kt++)
        am = wmma_bf16(load_a_frag(&h2s[0][0], 512, 0, kt * 32), load_b_frag(BWmel, 5, kt, W), am);
      if (col < NMELS){
#pragma unroll
        for (int r = 0; r < 8; r++){
          int b = bb + r;
          outMel[(size_t)b * NMELS * LMEL + (size_t)col * LMEL + step] = am[r];
          pmel[(size_t)(b * ROWS_PN + step + 2) * 96 + col] = f2bf(am[r]);
        }
      }
    } else if (W == 5 && lane < 16){
      int b = lane;
      float s = 0.f;
      for (int d = 0; d < 512; d++) s += bf2f(h2s[b][d]) * Wstop[d];
      outStop[b * LMEL + step] = sigmf(s + bstop[0]);
    }
    __syncthreads();
  }
}

// ================= host side =================
extern "C" void kernel_launch(void* const* d_in, const int* in_sizes, int n_in,
                              void* d_out, int out_size, void* d_ws, size_t ws_size,
                              hipStream_t stream){
  (void)in_sizes; (void)n_in; (void)out_size;
  const int*   text = (const int*)d_in[0];
  const float* melT = (const float*)d_in[1];
  const float* emb  = (const float*)d_in[2];
  const float* convW[3], *convB[3], *convG[3], *convBe[3];
  for (int l = 0; l < 3; l++){
    convW[l] = (const float*)d_in[3 + 4*l]; convB[l] = (const float*)d_in[4 + 4*l];
    convG[l] = (const float*)d_in[5 + 4*l]; convBe[l] = (const float*)d_in[6 + 4*l];
  }
  const float* wihF = (const float*)d_in[15], *whhF = (const float*)d_in[16];
  const float* bihF = (const float*)d_in[17], *bhhF = (const float*)d_in[18];
  const float* wihB = (const float*)d_in[19], *whhB = (const float*)d_in[20];
  const float* bihB = (const float*)d_in[21], *bhhB = (const float*)d_in[22];
  const float* preW1 = (const float*)d_in[23], *preB1 = (const float*)d_in[24];
  const float* preW2 = (const float*)d_in[25], *preB2 = (const float*)d_in[26];
  const float* attIh = (const float*)d_in[27], *attHh = (const float*)d_in[28];
  const float* attBi = (const float*)d_in[29], *attBh = (const float*)d_in[30];
  const float* decIh = (const float*)d_in[31], *decHh = (const float*)d_in[32];
  const float* decBi = (const float*)d_in[33], *decBh = (const float*)d_in[34];
  const float* Wq = (const float*)d_in[35], *Wk = (const float*)d_in[36];
  const float* Wv = (const float*)d_in[37], *Wl = (const float*)d_in[38];
  const float* locW = (const float*)d_in[39], *locB = (const float*)d_in[40];
  const float* wscore = (const float*)d_in[41];
  const float* Wmel = (const float*)d_in[42], *bmel = (const float*)d_in[43];
  const float* Wstop = (const float*)d_in[44], *bstop = (const float*)d_in[45];
  const float* pnW[5], *pnB[5], *pnG[4], *pnBe[4];
  for (int l = 0; l < 4; l++){
    pnW[l] = (const float*)d_in[46 + 4*l]; pnB[l] = (const float*)d_in[47 + 4*l];
    pnG[l] = (const float*)d_in[48 + 4*l]; pnBe[l] = (const float*)d_in[49 + 4*l];
  }
  pnW[4] = (const float*)d_in[62]; pnB[4] = (const float*)d_in[63];

  float* outMel  = (float*)d_out;
  float* outStop = outMel + (size_t)BATCH * NMELS * LMEL;
  float* outAttn = outStop + (size_t)BATCH * LMEL;

  // ---- workspace bump allocator ----
  char* base = (char*)d_ws; size_t off = 0;
  auto take = [&](size_t bytes)->void*{
    off = (off + 255) & ~(size_t)255; void* p = base + off; off += bytes; return p; };
  auto takeBF = [&](size_t n)->bf16*{ return (bf16*)take(n * 2); };
  auto takeF  = [&](size_t n)->float*{ return (float*)take(n * 4); };

  bf16* Bconv   = takeBF((size_t)3 * 5 * 8 * 16 * 512);      // enc conv taps
  bf16* BihF    = takeBF((size_t)8 * 128 * 512);
  bf16* BhhF    = takeBF((size_t)16 * 128 * 512);
  bf16* BihB    = takeBF((size_t)8 * 128 * 512);
  bf16* BhhB    = takeBF((size_t)16 * 128 * 512);
  bf16* Bpre1   = takeBF((size_t)3 * 32 * 512);
  bf16* Bpre2   = takeBF((size_t)16 * 32 * 512);
  bf16* BattPre = takeBF((size_t)16 * 128 * 512);
  bf16* BattCtx = takeBF((size_t)16 * 128 * 512);
  bf16* BattHh  = takeBF((size_t)16 * 128 * 512);
  bf16* BdecIh  = takeBF((size_t)32 * 128 * 512);
  bf16* BdecHh  = takeBF((size_t)16 * 128 * 512);
  bf16* BWq     = takeBF((size_t)16 * 8 * 512);
  bf16* BWk     = takeBF((size_t)32 * 8 * 512);
  bf16* BWv     = takeBF((size_t)32 * 32 * 512);
  bf16* BM2     = takeBF((size_t)1 * 8 * 512);
  bf16* BWmel   = takeBF((size_t)16 * 5 * 512);
  bf16* Bpn0    = takeBF((size_t)5 * 3 * 32 * 512);
  bf16* Bpn1    = takeBF((size_t)5 * 16 * 32 * 512);
  bf16* Bpn2    = takeBF((size_t)5 * 16 * 32 * 512);
  bf16* Bpn3    = takeBF((size_t)5 * 16 * 32 * 512);
  bf16* Bpn4    = takeBF((size_t)5 * 16 * 5 * 512);
  bf16* xpad0   = takeBF((size_t)BATCH * ROWS_ENC * EMB);
  bf16* xpad1   = takeBF((size_t)BATCH * ROWS_ENC * EMB);
  bf16* xsA     = takeBF((size_t)TTEXT * BATCH * EMB);
  bf16* encA    = takeBF((size_t)BATCH * TTEXT * 1024);
  bf16* prevA   = takeBF((size_t)LMEL * BATCH * 96);
  bf16* preh    = takeBF((size_t)LMEL * BATCH * 512);
  bf16* preA    = takeBF((size_t)LMEL * BATCH * 512);
  bf16* pmel    = takeBF((size_t)BATCH * ROWS_PN * 96);
  bf16* ppad0   = takeBF((size_t)BATCH * ROWS_PN * 512);
  bf16* ppad1   = takeBF((size_t)BATCH * ROWS_PN * 512);
  bf16* ctxG    = takeBF((size_t)16 * 512);
  float* xprojF = takeF((size_t)TTEXT * BATCH * 2048);
  float* xprojB = takeF((size_t)TTEXT * BATCH * 2048);
  float* preproj= takeF((size_t)LMEL * BATCH * 2048);
  float* keys   = takeF((size_t)BATCH * TTEXT * 128);
  float* encV   = takeF((size_t)BATCH * TTEXT * 512);
  float* biasF2 = takeF(2048);
  float* biasB2 = takeF(2048);
  float* biasA2 = takeF(2048);
  float* biasD2 = takeF(2048);
  float* M2f    = takeF(128 * 31);
  float* cbiasF = takeF(128);
  (void)ws_size;

  // zero whole used workspace (gives zero padding everywhere)
  hipMemsetAsync(d_ws, 0, off, stream);

  auto cvt = [&](const float* src, bf16* dst, int N, int K, int sN, int sK, int o, int Kp){
    size_t total = (size_t)(Kp >> 5) * (N >> 4) * 512;
    int blocks = (int)CDIV(total, 256); if (blocks > 8192) blocks = 8192;
    hipLaunchKernelGGL(k_convert_b, dim3(blocks), dim3(256), 0, stream, src, dst, N, K, sN, sK, o, Kp);
  };
  auto gemm = [&](const bf16* A, int lda, const bf16* B, const float* bias,
                  float* Cf, bf16* Cb, int M, int N, int K, int relu){
    int Nt = N / 16;
    if ((Nt & 3) == 0){
      int tiles = (M / 16) * (Nt / 4);
      int blocks = CDIV(tiles, 8); if (blocks > 4096) blocks = 4096;
      hipLaunchKernelGGL((k_gemm<4>), dim3(blocks), dim3(256), 0, stream, A, lda, B, bias, Cf, Cb, M, N, K, relu);
    } else {
      int tiles = (M / 16) * Nt;
      int blocks = CDIV(tiles, 8); if (blocks > 4096) blocks = 4096;
      hipLaunchKernelGGL((k_gemm<1>), dim3(blocks), dim3(256), 0, stream, A, lda, B, bias, Cf, Cb, M, N, K, relu);
    }
  };
  auto conv5 = [&](const bf16* X, int rowsB, int ldx, const bf16* Wsw, int Kp, int Coutp, int Cout,
                   const float* cb, const float* g, const float* be,
                   bf16* Y, int ldy, float* Yres, int Tvalid, int relu){
    int Ttiles = CDIV(Tvalid, 16);
    int Nt = Coutp / 16;
    if ((Nt & 3) == 0){
      int tiles = BATCH * Ttiles * (Nt / 4);
      int blocks = CDIV(tiles, 8); if (blocks > 4096) blocks = 4096;
      hipLaunchKernelGGL((k_conv5<4>), dim3(blocks), dim3(256), 0, stream, X, rowsB, ldx, Wsw, Kp,
                         Coutp, Cout, cb, g, be, Y, ldy, Yres, Tvalid, Ttiles, relu);
    } else {
      int tiles = BATCH * Ttiles * Nt;
      int blocks = CDIV(tiles, 8); if (blocks > 4096) blocks = 4096;
      hipLaunchKernelGGL((k_conv5<1>), dim3(blocks), dim3(256), 0, stream, X, rowsB, ldx, Wsw, Kp,
                         Coutp, Cout, cb, g, be, Y, ldy, Yres, Tvalid, Ttiles, relu);
    }
  };

  // ---- weight conversion ----
  for (int l = 0; l < 3; l++)
    for (int tap = 0; tap < 5; tap++)
      cvt(convW[l], Bconv + ((size_t)(l * 5 + tap) * 8 * 16 * 512), 256, 256, 256 * 5, 5, tap, 256);
  cvt(wihF, BihF, 2048, 256, 256, 1, 0, 256);
  cvt(whhF, BhhF, 2048, 512, 512, 1, 0, 512);
  cvt(wihB, BihB, 2048, 256, 256, 1, 0, 256);
  cvt(whhB, BhhB, 2048, 512, 512, 1, 0, 512);
  cvt(preW1, Bpre1, 512, 80, 80, 1, 0, 96);
  cvt(preW2, Bpre2, 512, 512, 512, 1, 0, 512);
  cvt(attIh, BattPre, 2048, 512, 1024, 1, 0, 512);
  cvt(attIh, BattCtx, 2048, 512, 1024, 1, 512, 512);
  cvt(attHh, BattHh, 2048, 512, 512, 1, 0, 512);
  cvt(decIh, BdecIh, 2048, 1024, 1024, 1, 0, 1024);
  cvt(decHh, BdecHh, 2048, 512, 512, 1, 0, 512);
  cvt(Wq, BWq, 128, 512, 512, 1, 0, 512);
  cvt(Wk, BWk, 128, 1024, 1024, 1, 0, 1024);
  cvt(Wv, BWv, 512, 1024, 1024, 1, 0, 1024);
  cvt(Wmel, BWmel, 80, 512, 512, 1, 0, 512);
  for (int tap = 0; tap < 5; tap++){
    cvt(pnW[0], Bpn0 + ((size_t)tap * 3 * 32 * 512), 512, 80, 80 * 5, 5, tap, 96);
    cvt(pnW[1], Bpn1 + ((size_t)tap * 16 * 32 * 512), 512, 512, 512 * 5, 5, tap, 512);
    cvt(pnW[2], Bpn2 + ((size_t)tap * 16 * 32 * 512), 512, 512, 512 * 5, 5, tap, 512);
    cvt(pnW[3], Bpn3 + ((size_t)tap * 16 * 32 * 512), 512, 512, 512 * 5, 5, tap, 512);
    cvt(pnW[4], Bpn4 + ((size_t)tap * 16 * 5 * 512), 80, 512, 512 * 5, 5, tap, 512);
  }
  hipLaunchKernelGGL(k_m2, dim3(CDIV(128 * 31, 128)), dim3(128), 0, stream, Wl, locW, locB, M2f, cbiasF);
  cvt(M2f, BM2, 128, 31, 31, 1, 0, 32);
  hipLaunchKernelGGL(k_vecadd, dim3(8), dim3(256), 0, stream, bihF, bhhF, biasF2, 2048);
  hipLaunchKernelGGL(k_vecadd, dim3(8), dim3(256), 0, stream, bihB, bhhB, biasB2, 2048);
  hipLaunchKernelGGL(k_vecadd, dim3(8), dim3(256), 0, stream, attBi, attBh, biasA2, 2048);
  hipLaunchKernelGGL(k_vecadd, dim3(8), dim3(256), 0, stream, decBi, decBh, biasD2, 2048);

  // ---- encoder: embedding -> 3x(conv+BN+relu) -> BiLSTM ----
  hipLaunchKernelGGL(k_embed, dim3(CDIV(BATCH * TTEXT * EMB, 256)), dim3(256), 0, stream, text, emb, xpad0);
  conv5(xpad0, ROWS_ENC, 256, Bconv + 0 * 5 * 8 * 16 * 512, 256, 256, 256,
        convB[0], convG[0], convBe[0], xpad1, 256, nullptr, TTEXT, 1);
  conv5(xpad1, ROWS_ENC, 256, Bconv + (size_t)1 * 5 * 8 * 16 * 512, 256, 256, 256,
        convB[1], convG[1], convBe[1], xpad0, 256, nullptr, TTEXT, 1);
  conv5(xpad0, ROWS_ENC, 256, Bconv + (size_t)2 * 5 * 8 * 16 * 512, 256, 256, 256,
        convB[2], convG[2], convBe[2], xpad1, 256, nullptr, TTEXT, 1);
  hipLaunchKernelGGL(k_repack, dim3(100), dim3(256), 0, stream, xpad1, xsA);
  gemm(xsA, 256, BihF, biasF2, xprojF, nullptr, TTEXT * BATCH, 2048, 256, 0);
  gemm(xsA, 256, BihB, biasB2, xprojB, nullptr, TTEXT * BATCH, 2048, 256, 0);
  hipLaunchKernelGGL(k_lstm_enc, dim3(2), dim3(1024), 0, stream, xprojF, xprojB, BhhF, BhhB, encA);
  gemm(encA, 1024, BWk, nullptr, keys, nullptr, BATCH * TTEXT, 128, 1024, 0);
  gemm(encA, 1024, BWv, nullptr, encV, nullptr, BATCH * TTEXT, 512, 1024, 0);

  // ---- prenet + hoisted att_lstm input projection ----
  hipLaunchKernelGGL(k_prev, dim3(CDIV(LMEL * BATCH * 96, 256)), dim3(256), 0, stream, melT, prevA);
  gemm(prevA, 96, Bpre1, preB1, nullptr, preh, LMEL * BATCH, 512, 96, 1);
  gemm(preh, 512, Bpre2, preB2, nullptr, preA, LMEL * BATCH, 512, 512, 1);
  gemm(preA, 512, BattPre, biasA2, preproj, nullptr, LMEL * BATCH, 2048, 512, 0);

  // ---- persistent attention decoder ----
  hipLaunchKernelGGL(k_decoder, dim3(1), dim3(1024), 0, stream,
                     preproj, BattCtx, BattHh, BdecIh, BdecHh, biasD2, BWq, BM2, cbiasF,
                     keys, wscore, encV, BWmel, bmel, Wstop, bstop,
                     ctxG, pmel, outMel, outStop, outAttn);

  // ---- postnet (residual into outMel) ----
  conv5(pmel, ROWS_PN, 96, Bpn0, 96, 512, 512, pnB[0], pnG[0], pnBe[0], ppad0, 512, nullptr, LMEL, 1);
  conv5(ppad0, ROWS_PN, 512, Bpn1, 512, 512, 512, pnB[1], pnG[1], pnBe[1], ppad1, 512, nullptr, LMEL, 1);
  conv5(ppad1, ROWS_PN, 512, Bpn2, 512, 512, 512, pnB[2], pnG[2], pnBe[2], ppad0, 512, nullptr, LMEL, 1);
  conv5(ppad0, ROWS_PN, 512, Bpn3, 512, 512, 512, pnB[3], pnG[3], pnBe[3], ppad1, 512, nullptr, LMEL, 1);
  conv5(ppad1, ROWS_PN, 512, Bpn4, 512, 80, 80, pnB[4], nullptr, nullptr, nullptr, 0, outMel, LMEL, 0);
}
